// DoubleStreamMixerProcessor_18227841204555
// MI455X (gfx1250) — matrix operand order repaired and checked by reading, not compiled, not run on hardware
//
#include <hip/hip_runtime.h>

typedef __attribute__((ext_vector_type(16))) __bf16 v16bf;
typedef __attribute__((ext_vector_type(8)))  float  v8f;

#define D2048 2048
#define NSTREAM_TXT 256
#define LTOT 1280

static __device__ inline v8f v8f_zero() {
  v8f v;
#pragma unroll
  for (int i = 0; i < 8; ++i) v[i] = 0.f;
  return v;
}

static __device__ inline v8f wmma_bf16(v16bf a, v16bf b, v8f c) {
  return __builtin_amdgcn_wmma_f32_16x16x32_bf16(false, a, false, b, (short)0, c,
                                                 false, false);
}

// CDNA5 async HBM->LDS copy (16B per lane), tracked by ASYNCcnt.
// LDS destination VGPR = low 32 bits of the generic pointer (flat aperture
// truncation: LDS address is the low 32 bits, ISA 10.2).
static __device__ inline void async_cp16(const float* g, float* lds) {
  unsigned loff = (unsigned)(unsigned long long)lds;
  asm volatile("global_load_async_to_lds_b128 %0, %1, off"
               :: "v"(loff), "v"(g) : "memory");
}
static __device__ inline void wait_async4() {
  asm volatile("s_wait_asynccnt 0x4" ::: "memory");
}
static __device__ inline void wait_async0() {
  asm volatile("s_wait_asynccnt 0x0" ::: "memory");
}

// ---------------------------------------------------------------- silu(vec)
__global__ void silu_kernel(const float* __restrict__ vec, float* __restrict__ vact) {
  int i = blockIdx.x * 256 + threadIdx.x;
  if (i < D2048) {
    float v = vec[i];
    vact[i] = v / (1.f + __expf(-v));
  }
}

// ------------------------------------------------ modulation GEMV (both streams)
__global__ __launch_bounds__(256) void mod_kernel(
    const float* __restrict__ vact,
    const float* __restrict__ wI, const float* __restrict__ bI,
    const float* __restrict__ wT, const float* __restrict__ bT,
    float* __restrict__ imod, float* __restrict__ tmod) {
  int j = blockIdx.x * 256 + threadIdx.x;  // 0..24575
  bool isI = j < 12288;
  int col = isI ? j : j - 12288;
  const float* w = isI ? wI : wT;
  const float* b = isI ? bI : bT;
  float s = b[col];
  for (int k = 0; k < D2048; ++k) s += vact[k] * w[(size_t)k * 12288 + col];
  (isI ? imod : tmod)[col] = s;
}

// ----------------------------------------- layernorm + (1+scale)*x + shift
__global__ __launch_bounds__(256) void ln_mod_kernel(
    const float* __restrict__ xt, const float* __restrict__ xi,
    const float* __restrict__ modt, const float* __restrict__ modi,
    int shChunk, int scChunk, float* __restrict__ out) {
  int l = blockIdx.x;
  const float* x = (l < NSTREAM_TXT) ? xt + (size_t)l * D2048
                                     : xi + (size_t)(l - NSTREAM_TXT) * D2048;
  const float* mo = (l < NSTREAM_TXT) ? modt : modi;
  float s = 0.f, s2 = 0.f;
  for (int c = threadIdx.x; c < D2048; c += 256) {
    float v = x[c];
    s += v;
    s2 += v * v;
  }
#pragma unroll
  for (int m = 16; m >= 1; m >>= 1) {
    s += __shfl_xor(s, m, 32);
    s2 += __shfl_xor(s2, m, 32);
  }
  __shared__ float rs[8], rs2[8];
  int wave = threadIdx.x >> 5, lane = threadIdx.x & 31;
  if (lane == 0) { rs[wave] = s; rs2[wave] = s2; }
  __syncthreads();
  s = 0.f; s2 = 0.f;
#pragma unroll
  for (int w = 0; w < 8; ++w) { s += rs[w]; s2 += rs2[w]; }
  float mu = s * (1.f / 2048.f);
  float var = s2 * (1.f / 2048.f) - mu * mu;
  float rstd = rsqrtf(var + 1e-6f);
  const float* sh = mo + shChunk * D2048;
  const float* sc = mo + scChunk * D2048;
  for (int c = threadIdx.x; c < D2048; c += 256)
    out[(size_t)l * D2048 + c] = (1.f + sc[c]) * ((x[c] - mu) * rstd) + sh[c];
}

// ------------------------------------------------- LoRA down-proj: A @ la (rank 16)
__global__ __launch_bounds__(256) void lora16_kernel(
    const float* __restrict__ A, const float* __restrict__ lat,
    const float* __restrict__ lai, float* __restrict__ out, int K) {
  int l = blockIdx.x;
  const float* la = (l < NSTREAM_TXT) ? lat : lai;
  const float* ar = A + (size_t)l * K;
  float p[16];
#pragma unroll
  for (int r = 0; r < 16; ++r) p[r] = 0.f;
  for (int k = threadIdx.x; k < K; k += 256) {
    float av = ar[k];
#pragma unroll
    for (int r = 0; r < 16; ++r) p[r] += av * la[(size_t)k * 16 + r];
  }
#pragma unroll
  for (int r = 0; r < 16; ++r)
#pragma unroll
    for (int m = 16; m >= 1; m >>= 1) p[r] += __shfl_xor(p[r], m, 32);
  __shared__ float red[8][16];
  int wave = threadIdx.x >> 5, lane = threadIdx.x & 31;
  if (lane == 0)
#pragma unroll
    for (int r = 0; r < 16; ++r) red[wave][r] = p[r];
  __syncthreads();
  if (threadIdx.x < 16) {
    float s = 0.f;
#pragma unroll
    for (int w = 0; w < 8; ++w) s += red[w][threadIdx.x];
    out[(size_t)l * 16 + threadIdx.x] = s;
  }
}

// ------------------------- qkv -> rms-norm + rope + bf16 pack into [H][L][HD]
__global__ __launch_bounds__(256) void rmsrope_kernel(
    const float* __restrict__ qkv,
    const float* __restrict__ qst, const float* __restrict__ kst,
    const float* __restrict__ qsi, const float* __restrict__ ksi,
    const float* __restrict__ pe,
    __bf16* __restrict__ qb, __bf16* __restrict__ kbuf, __bf16* __restrict__ vbuf) {
  int wid = (blockIdx.x << 3) + (threadIdx.x >> 5);  // 0..20479
  int lane = threadIdx.x & 31;
  int h = wid & 15, l = wid >> 4;
  const float* qs = (l < NSTREAM_TXT) ? qst : qsi;
  const float* ks = (l < NSTREAM_TXT) ? kst : ksi;
  const float* src = qkv + (size_t)l * 6144 + h * 128;
  __bf16* dstq = qb + ((size_t)h * LTOT + l) * 128;
  __bf16* dstk = kbuf + ((size_t)h * LTOT + l) * 128;
  __bf16* dstv = vbuf + ((size_t)h * LTOT + l) * 128;
  const float* per = pe + (size_t)l * 256;  // 64*2*2 per position

#pragma unroll
  for (int s = 0; s < 2; ++s) {
    const float* x = src + s * D2048;
    const float* scl = s ? ks : qs;
    __bf16* dst = s ? dstk : dstq;
    float x0a = x[2 * lane], x1a = x[2 * lane + 1];
    float x0b = x[2 * (lane + 32)], x1b = x[2 * (lane + 32) + 1];
    float ss = x0a * x0a + x1a * x1a + x0b * x0b + x1b * x1b;
#pragma unroll
    for (int m = 16; m >= 1; m >>= 1) ss += __shfl_xor(ss, m, 32);
    float rr = rsqrtf(ss * (1.f / 128.f) + 1e-6f);
    {
      int i = lane;
      float n0 = x0a * rr * scl[2 * i], n1 = x1a * rr * scl[2 * i + 1];
      const float* pp = per + i * 4;
      dst[2 * i] = (__bf16)(pp[0] * n0 + pp[1] * n1);
      dst[2 * i + 1] = (__bf16)(pp[2] * n0 + pp[3] * n1);
    }
    {
      int i = lane + 32;
      float n0 = x0b * rr * scl[2 * i], n1 = x1b * rr * scl[2 * i + 1];
      const float* pp = per + i * 4;
      dst[2 * i] = (__bf16)(pp[0] * n0 + pp[1] * n1);
      dst[2 * i + 1] = (__bf16)(pp[2] * n0 + pp[3] * n1);
    }
  }
  const float* xv = src + 2 * D2048;
#pragma unroll
  for (int j = 0; j < 4; ++j) {
    int c = lane + 32 * j;
    dstv[c] = (__bf16)xv[c];
  }
}

// -------------------------------- flash attention: one wave per (head, 16-q tile)
__global__ __launch_bounds__(256) void attn_kernel(
    const __bf16* __restrict__ qb, const __bf16* __restrict__ kbuf,
    const __bf16* __restrict__ vbuf, float* __restrict__ aout) {
  __shared__ __bf16 Pld[8][16 * 34];
  int wave = threadIdx.x >> 5, lane = threadIdx.x & 31;
  int tile = blockIdx.x * 8 + wave;  // 0..1279
  int h = tile / 80, qt = tile % 80;
  int q0 = qt * 16;
  int r = lane & 15;
  int kb_ = (lane < 16) ? 0 : 8;   // unused; A-fragment K base is 0/16 (half-wave)
  int hb = (lane < 16) ? 0 : 8;    // C-layout row base (half-wave)
  const __bf16* Q = qb + ((size_t)h * LTOT + q0) * 128;

  v16bf qf[4];
#pragma unroll
  for (int f = 0; f < 4; ++f)
#pragma unroll
    for (int e = 0; e < 16; ++e) {
      int kk = f * 32 + ((lane < 16) ? 0 : 16) + e;
      qf[f][e] = Q[(size_t)r * 128 + kk];
    }

  v8f o[8];
#pragma unroll
  for (int nt = 0; nt < 8; ++nt) o[nt] = v8f_zero();
  float mrow[8], lrow[8];
#pragma unroll
  for (int e = 0; e < 8; ++e) { mrow[e] = -1e30f; lrow[e] = 0.f; }
  const float scl = 0.08838834764831845f;  // 1/sqrt(128)

  for (int s0 = 0; s0 < LTOT; s0 += 32) {
    v8f sa0 = v8f_zero(), sa1 = v8f_zero();
    const __bf16* K0 = kbuf + ((size_t)h * LTOT + s0 + r) * 128;
    const __bf16* K1 = kbuf + ((size_t)h * LTOT + s0 + 16 + r) * 128;
    if (s0 + 32 < LTOT) {  // warm L2/L0 for next K/V tiles
      __builtin_prefetch(kbuf + ((size_t)h * LTOT + s0 + 32 + r) * 128, 0, 3);
      __builtin_prefetch(vbuf + ((size_t)h * LTOT + s0 + 32 + r) * 128, 0, 3);
    }
#pragma unroll
    for (int f = 0; f < 4; ++f) {
      v16bf b0, b1;
      int cb = f * 32 + ((lane < 16) ? 0 : 16);
#pragma unroll
      for (int e = 0; e < 16; ++e) {
        b0[e] = K0[cb + e];
        b1[e] = K1[cb + e];
      }
      sa0 = wmma_bf16(qf[f], b0, sa0);
      sa1 = wmma_bf16(qf[f], b1, sa1);
    }
    // online softmax over this 16x32 S block (rows live per half-wave)
    float cf[8];
#pragma unroll
    for (int e = 0; e < 8; ++e) {
      float v0 = sa0[e] * scl, v1 = sa1[e] * scl;
      float mx = fmaxf(v0, v1);
#pragma unroll
      for (int m = 8; m >= 1; m >>= 1) mx = fmaxf(mx, __shfl_xor(mx, m, 32));
      float nm = fmaxf(mrow[e], mx);
      float p0 = __expf(v0 - nm), p1 = __expf(v1 - nm);
      float ps = p0 + p1;
#pragma unroll
      for (int m = 8; m >= 1; m >>= 1) ps += __shfl_xor(ps, m, 32);
      float c = __expf(mrow[e] - nm);
      lrow[e] = lrow[e] * c + ps;
      mrow[e] = nm;
      cf[e] = c;
      Pld[wave][(e + hb) * 34 + r] = (__bf16)p0;
      Pld[wave][(e + hb) * 34 + 16 + r] = (__bf16)p1;
    }
#pragma unroll
    for (int nt = 0; nt < 8; ++nt)
#pragma unroll
      for (int e = 0; e < 8; ++e) o[nt][e] *= cf[e];
    __syncthreads();
    v16bf pf;
#pragma unroll
    for (int e = 0; e < 16; ++e) {
      int kk = ((lane < 16) ? 0 : 16) + e;
      pf[e] = Pld[wave][r * 34 + kk];
    }
    int kk0 = (lane < 16) ? 0 : 16;
#pragma unroll
    for (int nt = 0; nt < 8; ++nt) {
      v16bf bv;
#pragma unroll
      for (int e = 0; e < 16; ++e)
        bv[e] = vbuf[((size_t)h * LTOT + s0 + kk0 + e) * 128 + nt * 16 + r];
      o[nt] = wmma_bf16(pf, bv, o[nt]);
    }
    __syncthreads();
  }
#pragma unroll
  for (int nt = 0; nt < 8; ++nt)
#pragma unroll
    for (int e = 0; e < 8; ++e) {
      int rowq = q0 + e + hb;
      aout[(size_t)rowq * D2048 + h * 128 + nt * 16 + r] = o[nt][e] / lrow[e];
    }
}

// -------------------------------------------------------- tiled WMMA GEMM
// Async double-buffered HBM->LDS staging (ASYNCcnt ping-pong), fp32 tiles in
// LDS, f32->bf16 conversion fused into fragment assembly.
// MODE 0: qkv   : Cdir = A@W + bias + LT@LB
// MODE 1: proj  : Cout = X + g1*(A@W + bias + LT@LB)   (writes d_out)
// MODE 2: mlp1  : Cdir = gelu_tanh(A@W + bias)
// MODE 3: mlp2  : Cout += g2*(A@W + bias)              (accumulates d_out)
#define AS_STRIDE 36  // multiple of 4 floats: keeps 16B alignment for b128
#define BS_STRIDE 68
template <int MODE>
__global__ __launch_bounds__(256) void gemm_bf16(
    const float* __restrict__ A,
    const float* __restrict__ Wt, const float* __restrict__ Wi,
    const float* __restrict__ Bt, const float* __restrict__ Bi,
    const float* __restrict__ LT,
    const float* __restrict__ LBt, const float* __restrict__ LBi,
    const float* __restrict__ MODt, const float* __restrict__ MODi,
    const float* __restrict__ Xt, const float* __restrict__ Xi,
    float* __restrict__ Cdir, float* __restrict__ Cout,
    int M, int N, int K) {
  __shared__ float As[2][64 * AS_STRIDE];
  __shared__ float Bs[2][32 * BS_STRIDE];
  int n0 = blockIdx.x * 64, row0 = blockIdx.y * 64;
  bool is_txt = row0 < NSTREAM_TXT;
  const float* W = is_txt ? Wt : Wi;
  int tid = threadIdx.x, lane = tid & 31, wave = tid >> 5;
  int wr = wave >> 2, wc = wave & 3;
  v8f acc0 = v8f_zero(), acc1 = v8f_zero();

  // per-thread staging coordinates (4 async b128 per thread per K-tile)
  int idx4a = tid << 2, idx4b = (tid + 256) << 2;
  int ar0 = idx4a >> 5, ac0 = idx4a & 31, ar1 = idx4b >> 5, ac1 = idx4b & 31;
  int br0 = idx4a >> 6, bc0 = idx4a & 63, br1 = idx4b >> 6, bc1 = idx4b & 63;

  auto stage = [&](int k0, int buf) {
    async_cp16(A + (size_t)(row0 + ar0) * K + k0 + ac0, &As[buf][ar0 * AS_STRIDE + ac0]);
    async_cp16(W + (size_t)(k0 + br0) * N + n0 + bc0, &Bs[buf][br0 * BS_STRIDE + bc0]);
    async_cp16(A + (size_t)(row0 + ar1) * K + k0 + ac1, &As[buf][ar1 * AS_STRIDE + ac1]);
    async_cp16(W + (size_t)(k0 + br1) * N + n0 + bc1, &Bs[buf][br1 * BS_STRIDE + bc1]);
  };

  stage(0, 0);
  int nk = K >> 5;
  int r0 = wr * 32 + (lane & 15);
  int kb_ = (lane < 16) ? 0 : 8;
  int cb = (lane < 16) ? 0 : 16;
  int nn = wc * 16 + (lane & 15);
  for (int t = 0; t < nk; ++t) {
    if (t + 1 < nk) {
      stage((t + 1) << 5, (t + 1) & 1);
      wait_async4();  // oldest 4 (tile t) have landed in LDS
    } else {
      wait_async0();
    }
    __syncthreads();
    const float* Ab = As[t & 1];
    const float* Bb = Bs[t & 1];
    v16bf af0, af1, bf;
#pragma unroll
    for (int e = 0; e < 16; ++e) {
      int kk = cb + e;
      af0[e] = (__bf16)Ab[r0 * AS_STRIDE + kk];
      af1[e] = (__bf16)Ab[(r0 + 16) * AS_STRIDE + kk];
      bf[e] = (__bf16)Bb[(cb + e) * BS_STRIDE + nn];
    }
    acc0 = wmma_bf16(af0, bf, acc0);
    acc1 = wmma_bf16(af1, bf, acc1);
    __syncthreads();
  }

  const float* Bv = is_txt ? Bt : Bi;
  const float* LB = is_txt ? LBt : LBi;
  const float* MO = is_txt ? MODt : MODi;
  int col = n0 + wc * 16 + (lane & 15);
  float bias = Bv[col];
#pragma unroll
  for (int half = 0; half < 2; ++half) {
    v8f acc = half ? acc1 : acc0;
    int rbase = row0 + wr * 32 + half * 16 + ((lane < 16) ? 0 : 8);
#pragma unroll
    for (int e = 0; e < 8; ++e) {
      int rowg = rbase + e;
      float v = acc[e] + bias;
      if (MODE == 0 || MODE == 1) {
        float lv = 0.f;
#pragma unroll
        for (int rr = 0; rr < 16; ++rr)
          lv += LT[(size_t)rowg * 16 + rr] * LB[(size_t)rr * N + col];
        v += lv;
      }
      if (MODE == 0) {
        Cdir[(size_t)rowg * N + col] = v;
      } else if (MODE == 1) {
        float g = MO[2 * D2048 + col];
        float x = is_txt ? Xt[(size_t)rowg * D2048 + col]
                         : Xi[(size_t)(rowg - NSTREAM_TXT) * D2048 + col];
        size_t off = is_txt ? ((size_t)1024 * D2048 + (size_t)rowg * D2048)
                            : ((size_t)(rowg - NSTREAM_TXT) * D2048);
        Cout[off + col] = x + g * v;
      } else if (MODE == 2) {
        float t = 0.7978845608028654f * (v + 0.044715f * v * v * v);
        Cdir[(size_t)rowg * N + col] = 0.5f * v * (1.0f + tanhf(t));
      } else {
        float g = MO[5 * D2048 + col];
        size_t off = is_txt ? ((size_t)1024 * D2048 + (size_t)rowg * D2048)
                            : ((size_t)(rowg - NSTREAM_TXT) * D2048);
        Cout[off + col] += g * v;
      }
    }
  }
}

// ----------------------------------------------------------------- launch
extern "C" void kernel_launch(void* const* d_in, const int* in_sizes, int n_in,
                              void* d_out, int out_size, void* d_ws, size_t ws_size,
                              hipStream_t stream) {
  (void)in_sizes; (void)n_in; (void)out_size; (void)ws_size;
  const float* img = (const float*)d_in[0];
  const float* txt = (const float*)d_in[1];
  const float* vec = (const float*)d_in[2];
  const float* pe  = (const float*)d_in[3];
  const float* IMG[16]; const float* TXT[16];
  for (int i = 0; i < 16; ++i) { IMG[i] = (const float*)d_in[4 + i];
                                 TXT[i] = (const float*)d_in[20 + i]; }
  enum { MOD_W, MOD_B, QKV_W, QKV_B, QKV_LA, QKV_LB, QS, KS,
         PROJ_W, PROJ_B, PROJ_LA, PROJ_LB, MLP_W1, MLP_B1, MLP_W2, MLP_B2 };

  float* ws = (float*)d_ws;
  constexpr size_t O_VACT  = 0;
  constexpr size_t O_IMOD  = O_VACT + 2048;
  constexpr size_t O_TMOD  = O_IMOD + 12288;
  constexpr size_t O_LORAT = O_TMOD + 12288;
  constexpr size_t O_LORAP = O_LORAT + 1280UL * 16;
  constexpr size_t O_M     = O_LORAP + 1280UL * 16;
  constexpr size_t O_QKV   = O_M + 1280UL * 2048;
  constexpr size_t O_A     = O_QKV + 1280UL * 6144;
  constexpr size_t O_H     = O_A + 1280UL * 2048;
  constexpr size_t O_QB    = O_H + 1280UL * 8192;
  constexpr size_t O_KB    = O_QB + (16UL * 1280 * 128) / 2;
  constexpr size_t O_VB    = O_KB + (16UL * 1280 * 128) / 2;

  float* vact = ws + O_VACT;
  float* imod = ws + O_IMOD;
  float* tmod = ws + O_TMOD;
  float* lorat = ws + O_LORAT;
  float* lorap = ws + O_LORAP;
  float* mbuf = ws + O_M;
  float* qkv = ws + O_QKV;
  float* abuf = ws + O_A;
  float* hbuf = ws + O_H;
  __bf16* qb = (__bf16*)(ws + O_QB);
  __bf16* kb = (__bf16*)(ws + O_KB);
  __bf16* vb = (__bf16*)(ws + O_VB);
  float* out = (float*)d_out;

  silu_kernel<<<8, 256, 0, stream>>>(vec, vact);
  mod_kernel<<<96, 256, 0, stream>>>(vact, IMG[MOD_W], IMG[MOD_B],
                                     TXT[MOD_W], TXT[MOD_B], imod, tmod);
  ln_mod_kernel<<<1280, 256, 0, stream>>>(txt, img, tmod, imod, 0, 1, mbuf);
  lora16_kernel<<<1280, 256, 0, stream>>>(mbuf, TXT[QKV_LA], IMG[QKV_LA], lorat, 2048);
  gemm_bf16<0><<<dim3(96, 20), 256, 0, stream>>>(
      mbuf, TXT[QKV_W], IMG[QKV_W], TXT[QKV_B], IMG[QKV_B],
      lorat, TXT[QKV_LB], IMG[QKV_LB], tmod, imod, txt, img,
      qkv, out, 1280, 6144, 2048);
  rmsrope_kernel<<<2560, 256, 0, stream>>>(qkv, TXT[QS], TXT[KS], IMG[QS], IMG[KS],
                                           pe, qb, kb, vb);
  attn_kernel<<<160, 256, 0, stream>>>(qb, kb, vb, abuf);
  lora16_kernel<<<1280, 256, 0, stream>>>(abuf, TXT[PROJ_LA], IMG[PROJ_LA], lorap, 2048);
  gemm_bf16<1><<<dim3(32, 20), 256, 0, stream>>>(
      abuf, TXT[PROJ_W], IMG[PROJ_W], TXT[PROJ_B], IMG[PROJ_B],
      lorap, TXT[PROJ_LB], IMG[PROJ_LB], tmod, imod, txt, img,
      qkv /*unused*/, out, 1280, 2048, 2048);
  ln_mod_kernel<<<1280, 256, 0, stream>>>(out + 1024UL * 2048, out, tmod, imod,
                                          3, 4, mbuf);
  gemm_bf16<2><<<dim3(128, 20), 256, 0, stream>>>(
      mbuf, TXT[MLP_W1], IMG[MLP_W1], TXT[MLP_B1], IMG[MLP_B1],
      lorat, TXT[QKV_LB], IMG[QKV_LB], tmod, imod, txt, img,
      hbuf, out, 1280, 8192, 2048);
  gemm_bf16<3><<<dim3(32, 20), 256, 0, stream>>>(
      hbuf, TXT[MLP_W2], IMG[MLP_W2], TXT[MLP_B2], IMG[MLP_B2],
      lorat, TXT[QKV_LB], IMG[QKV_LB], tmod, imod, txt, img,
      qkv /*unused*/, out, 1280, 2048, 8192);
}
